// SocialModel_50826642981008
// MI455X (gfx1250) — matrix operand order repaired and checked by reading, not compiled, not run on hardware
//
#include <hip/hip_runtime.h>

#define SSTEPS 32
#define NNODES 400
#define G2C    16
#define RR     128
#define EE     64
#define OUTC   5
#define KTILE  32
#define NKT    13      // ceil(400/32)
#define KPAD   416     // 13*32, zero-padded h columns

typedef __attribute__((ext_vector_type(16))) __bf16         v16bf;
typedef __attribute__((ext_vector_type(8)))  float          v8f;
typedef __attribute__((ext_vector_type(8)))  unsigned short ushort8;
typedef __attribute__((ext_vector_type(4)))  unsigned int   uint4v;
typedef __attribute__((ext_vector_type(4)))  int            int4v;
typedef __attribute__((ext_vector_type(8)))  int            int8v;

union BF16Frag { v16bf v; ushort8 h[2]; };

#if __has_builtin(__builtin_amdgcn_tensor_load_to_lds)
#define HAVE_TDM 1
#else
#define HAVE_TDM 0
#endif

__device__ __forceinline__ unsigned short f2bf(float f) {
  union { float f; unsigned int u; } c; c.f = f;
  unsigned int u = c.u;
  return (unsigned short)((u + 0x7FFFu + ((u >> 16) & 1u)) >> 16); // RNE
}

// ---------- once-per-launch prep ----------

// h0 [400][128] f32 -> h_bfT [128][416] bf16 (transposed, zero-padded cols)
__global__ void k_init(const float* __restrict__ h0, unsigned short* __restrict__ h_bfT) {
  int i = blockIdx.x * blockDim.x + threadIdx.x;
  if (i >= RR * KPAD) return;
  int r = i / KPAD, m = i % KPAD;
  float v = (m < NNODES) ? h0[(size_t)m * RR + r] : 0.f;
  h_bfT[i] = f2bf(v);
}

// wtT [64][2048] = W_t^T (bf16); wcatT [512][256] = [W_ih | W_hh] rows (bf16)
__global__ void k_wprep(const float* __restrict__ W_t, const float* __restrict__ W_ih,
                        const float* __restrict__ W_hh,
                        unsigned short* __restrict__ wtT, unsigned short* __restrict__ wcatT) {
  int i = blockIdx.x * blockDim.x + threadIdx.x;
  if (i < EE * 2048) {
    int e = i / 2048, k = i % 2048;
    wtT[i] = f2bf(W_t[(size_t)k * EE + e]);
  } else if (i < EE * 2048 + 512 * 256) {
    int q = i - EE * 2048;
    int j = q / 256, k = q % 256;
    float v = (k < RR) ? W_ih[(size_t)j * RR + k] : W_hh[(size_t)j * RR + (k - RR)];
    wcatT[q] = f2bf(v);
  }
}

// ---------- Tensor Data Mover: 2D tile (128 rows x 32 cols, 2B) from h_bfT ----------
__device__ __forceinline__ void tdm_load_htile(const unsigned short* gsrc, unsigned ldsaddr) {
#if HAVE_TDM
  unsigned long long ga = (unsigned long long)gsrc;
  uint4v g0;
  g0[0] = 1u;                                            // count=1, is_restore=0
  g0[1] = ldsaddr;                                       // lds_addr
  g0[2] = (unsigned)(ga & 0xFFFFFFFFull);                // global_addr[31:0]
  g0[3] = (unsigned)((ga >> 32) & 0x01FFFFFFull) | (2u << 30); // addr[56:32] | type=2
  int8v g1;
  g1[0] = 0x00010000;                                    // data_size=1 (2 bytes)
  g1[1] = (int)((unsigned)KPAD << 16);                   // tensor_dim0 = 416
  g1[2] = (int)((unsigned)RR << 16);                     // tensor_dim1 = 128
  g1[3] = (int)((unsigned)KTILE << 16);                  // tile_dim0 = 32
  g1[4] = RR;                                            // tile_dim1 = 128
  g1[5] = KPAD;                                          // tensor_dim0_stride = 416
  g1[6] = (int)((unsigned)(KPAD * RR) << 16);            // tensor_dim1_stride[15:0]
  g1[7] = (int)((unsigned)(KPAD * RR) >> 16);
  int4v z4 = {};
#if defined(__clang_major__) && (__clang_major__ >= 23)
  int8v z8 = {};
  __builtin_amdgcn_tensor_load_to_lds(g0, g1, z4, z4, z8, 0);
#else
  __builtin_amdgcn_tensor_load_to_lds(g0, g1, z4, z4, 0);
#endif
#else
  (void)gsrc; (void)ldsaddr;
#endif
}

// ---------- per step: social tensor GEMM  social[n] = grid[n]^T (16x400) @ h (400x128) ----------
__global__ __launch_bounds__(128)
void k_social(const float* __restrict__ grid_s,          // [400][400][16]
              const unsigned short* __restrict__ h_bfT,  // [128][416] bf16
              unsigned short* __restrict__ soc)          // [400][2048] bf16
{
  __shared__ __align__(16) unsigned short ldsA[16][32];   // A: [g][m_local]
  __shared__ __align__(16) unsigned short ldsB[128][32];  // B: [r][m_local]

  const int n    = blockIdx.x;
  const int t    = threadIdx.x;
  const int lane = t & 31;
  const int wave = t >> 5;
  const int half = lane >> 4;
  const int l16  = lane & 15;

  const float* gn = grid_s + (size_t)n * (NNODES * G2C);

  v8f acc0 = {};
  v8f acc1 = {};

  for (int kt = 0; kt < NKT; ++kt) {
#if HAVE_TDM
    if (wave == 0)
      tdm_load_htile(h_bfT + kt * KTILE, (unsigned)(unsigned long long)&ldsB[0][0]);
#else
    for (int q = t; q < 512; q += 128) {
      int r = q >> 2, c = (q & 3) << 3;
      *(ushort8*)&ldsB[r][c] = *(const ushort8*)(h_bfT + (size_t)r * KPAD + kt * KTILE + c);
    }
#endif
    // stage A: grid rows [32kt,+32) x 16g = 512 contiguous floats, convert+transpose
    {
      int mloc = t >> 2;
      int g0   = (t & 3) << 2;
      int m    = kt * KTILE + mloc;
      float4 v4 = make_float4(0.f, 0.f, 0.f, 0.f);
      if (m < NNODES) v4 = *(const float4*)(gn + (size_t)m * G2C + g0);
      ldsA[g0 + 0][mloc] = f2bf(v4.x);
      ldsA[g0 + 1][mloc] = f2bf(v4.y);
      ldsA[g0 + 2][mloc] = f2bf(v4.z);
      ldsA[g0 + 3][mloc] = f2bf(v4.w);
    }
    if (kt + 1 < NKT)
      __builtin_prefetch(gn + (size_t)(kt + 1) * 512 + t * 4, 0, 1);
#if HAVE_TDM
    if (wave == 0) __builtin_amdgcn_s_wait_tensorcnt(0);
#endif
    __syncthreads();

    BF16Frag afrag, b0, b1;
    {
      const ushort8* pa = (const ushort8*)&ldsA[l16][0];
      afrag.h[0] = pa[2 * half];
      afrag.h[1] = pa[2 * half + 1];
    }
    {
      const ushort8* pb = (const ushort8*)&ldsB[32 * wave + l16][0];
      b0.h[0] = pb[2 * half];
      b0.h[1] = pb[2 * half + 1];
      const ushort8* pc = (const ushort8*)&ldsB[32 * wave + 16 + l16][0];
      b1.h[0] = pc[2 * half];
      b1.h[1] = pc[2 * half + 1];
    }
    acc0 = __builtin_amdgcn_wmma_f32_16x16x32_bf16(false, afrag.v, false, b0.v,
                                                   (short)0, acc0, false, false);
    acc1 = __builtin_amdgcn_wmma_f32_16x16x32_bf16(false, afrag.v, false, b1.v,
                                                   (short)0, acc1, false, false);
    __syncthreads();
  }

  // store social tile as bf16: C layout lane l, vgpr v -> g = v + 8*(l/16), r = 32w + l%16
  {
    int r0 = 32 * wave;
    size_t base = (size_t)n * 2048;
#pragma unroll
    for (int v = 0; v < 8; ++v) {
      int g = v + 8 * half;
      soc[base + g * RR + r0 + l16]      = f2bf(acc0[v]);
      soc[base + g * RR + r0 + 16 + l16] = f2bf(acc1[v]);
    }
  }
}

// ---------- per step: embeddings + gates + LSTM pointwise, 16 nodes per block ----------
__global__ __launch_bounds__(128)
void k_dense(const unsigned short* __restrict__ soc,    // [400][2048] bf16
             const float* __restrict__ x_s,             // [400][2]
             const float* __restrict__ h_cur,           // [400][128] f32
             const float* __restrict__ c_cur,           // [400][128] f32
             const unsigned short* __restrict__ wtT,    // [64][2048] bf16
             const unsigned short* __restrict__ wcatT,  // [512][256] bf16
             const float* __restrict__ W_in, const float* __restrict__ b_in,
             const float* __restrict__ b_t,
             const float* __restrict__ b_ih, const float* __restrict__ b_hh,
             const float* __restrict__ W_out, const float* __restrict__ b_out,
             float* __restrict__ h_nxt, float* __restrict__ c_nxt,
             unsigned short* __restrict__ h_bfT,        // [128][416] for next step
             float* __restrict__ out_s)                 // [400][5]
{
  __shared__ __align__(16) unsigned short zh[16][256];  // [node][k] : emb | temb | h
  __shared__ float hnew[16][128];
  __shared__ float bsum[512];

  const int t    = threadIdx.x;
  const int lane = t & 31;
  const int wave = t >> 5;
  const int half = lane >> 4;
  const int l16  = lane & 15;
  const int nb   = blockIdx.x * 16;

  for (int j = t; j < 512; j += 128) bsum[j] = b_ih[j] + b_hh[j];

  // input embedding -> zh[:, 0:64]
  {
    int node = t >> 3;
    int e0   = (t & 7) * 8;
    float x0 = x_s[(size_t)(nb + node) * 2 + 0];
    float x1 = x_s[(size_t)(nb + node) * 2 + 1];
#pragma unroll
    for (int i = 0; i < 8; ++i) {
      int e = e0 + i;
      float v = x0 * W_in[e] + x1 * W_in[EE + e] + b_in[e];
      zh[node][e] = f2bf(v > 0.f ? v : 0.f);
    }
  }
  // h -> zh[:, 128:256]
  {
    int node = t >> 3;
    int r0   = (t & 7) * 16;
#pragma unroll
    for (int i = 0; i < 16; ++i)
      zh[node][RR + r0 + i] = f2bf(h_cur[(size_t)(nb + node) * RR + r0 + i]);
  }

  // tensor_emb GEMM: [16 x 2048] @ W_t -> [16 x 64]; wave w owns e-tile 16w
  v8f te = {};
  {
    const unsigned short* arow = soc + (size_t)(nb + l16) * 2048;
    const unsigned short* brow = wtT + (size_t)(16 * wave + l16) * 2048;
    for (int kt = 0; kt < 64; ++kt) {
      BF16Frag a, b;
      a.h[0] = *(const ushort8*)(arow + 32 * kt + 16 * half);
      a.h[1] = *(const ushort8*)(arow + 32 * kt + 16 * half + 8);
      b.h[0] = *(const ushort8*)(brow + 32 * kt + 16 * half);
      b.h[1] = *(const ushort8*)(brow + 32 * kt + 16 * half + 8);
      te = __builtin_amdgcn_wmma_f32_16x16x32_bf16(false, a.v, false, b.v,
                                                   (short)0, te, false, false);
    }
  }
  // relu(te + b_t) -> zh[:, 64:128]
  {
    int e = 16 * wave + l16;
    float bt = b_t[e];
#pragma unroll
    for (int v = 0; v < 8; ++v) {
      int node = v + 8 * half;
      float val = te[v] + bt;
      zh[node][EE + e] = f2bf(val > 0.f ? val : 0.f);
    }
  }
  __syncthreads();

  // gates GEMM: [16 x 256] @ [W_ih|W_hh]^T -> [16 x 512]; wave w owns col tiles w+4i
  v8f g8[8];
#pragma unroll
  for (int i = 0; i < 8; ++i) g8[i] = (v8f){};
  for (int kt = 0; kt < 8; ++kt) {
    BF16Frag a;
    a.h[0] = *(const ushort8*)&zh[l16][32 * kt + 16 * half];
    a.h[1] = *(const ushort8*)&zh[l16][32 * kt + 16 * half + 8];
#pragma unroll
    for (int i = 0; i < 8; ++i) {
      int j = 16 * (wave + 4 * i) + l16;
      BF16Frag b;
      const unsigned short* br = wcatT + (size_t)j * 256 + 32 * kt + 16 * half;
      b.h[0] = *(const ushort8*)br;
      b.h[1] = *(const ushort8*)(br + 8);
      g8[i] = __builtin_amdgcn_wmma_f32_16x16x32_bf16(false, a.v, false, b.v,
                                                      (short)0, g8[i], false, false);
    }
  }

  // LSTM pointwise: lane holds matching i/f/g/o (torch order) for (node, r)
#pragma unroll
  for (int i2 = 0; i2 < 2; ++i2) {
    int r = 16 * (wave + 4 * i2) + l16;
    float bi = bsum[r], bf = bsum[RR + r], bg = bsum[2 * RR + r], bo = bsum[3 * RR + r];
#pragma unroll
    for (int v = 0; v < 8; ++v) {
      int node = v + 8 * half;
      size_t gidx = (size_t)(nb + node) * RR + r;
      float ig = 1.f / (1.f + __expf(-(g8[i2][v] + bi)));
      float fg = 1.f / (1.f + __expf(-(g8[i2 + 2][v] + bf)));
      float gg = tanhf(g8[i2 + 4][v] + bg);
      float og = 1.f / (1.f + __expf(-(g8[i2 + 6][v] + bo)));
      float cn = fg * c_cur[gidx] + ig * gg;
      float hn = og * tanhf(cn);
      c_nxt[gidx] = cn;
      h_nxt[gidx] = hn;
      h_bfT[(size_t)r * KPAD + (nb + node)] = f2bf(hn);
      hnew[node][r] = hn;
    }
  }
  __syncthreads();

  // output projection: 16 nodes x 5 outputs
  if (t < 16 * OUTC) {
    int node = t / OUTC, o = t % OUTC;
    float acc = b_out[o];
    for (int k = 0; k < RR; ++k) acc += hnew[node][k] * W_out[k * OUTC + o];
    out_s[(size_t)(nb + node) * OUTC + o] = acc;
  }
}

extern "C" void kernel_launch(void* const* d_in, const int* in_sizes, int n_in,
                              void* d_out, int out_size, void* d_ws, size_t ws_size,
                              hipStream_t stream) {
  (void)in_sizes; (void)n_in; (void)out_size; (void)ws_size;
  const float* input_data = (const float*)d_in[0];   // [32,400,2]
  const float* grids      = (const float*)d_in[1];   // [32,400,400,16]
  const float* hidden0    = (const float*)d_in[2];
  const float* cell0      = (const float*)d_in[3];
  const float* W_in  = (const float*)d_in[4];
  const float* b_in  = (const float*)d_in[5];
  const float* W_t   = (const float*)d_in[6];
  const float* b_t   = (const float*)d_in[7];
  const float* W_ih  = (const float*)d_in[8];
  const float* b_ih  = (const float*)d_in[9];
  const float* W_hh  = (const float*)d_in[10];
  const float* b_hh  = (const float*)d_in[11];
  const float* W_out = (const float*)d_in[12];
  const float* b_out = (const float*)d_in[13];
  float* out = (float*)d_out;   // [32*400*5], then h_fin [400*128], c_fin [400*128]

  char* w = (char*)d_ws;
  float*          hA    = (float*)(w + 0);           // 204800 B
  float*          hB    = (float*)(w + 204800);
  float*          cA    = (float*)(w + 409600);
  float*          cB    = (float*)(w + 614400);
  unsigned short* hbfT  = (unsigned short*)(w + 819200);   // 128x416 bf16 = 106496 B
  unsigned short* soc   = (unsigned short*)(w + 925696);   // 400x2048 bf16 = 1638400 B
  unsigned short* wtT   = (unsigned short*)(w + 2564096);  // 64x2048 bf16 = 262144 B
  unsigned short* wcatT = (unsigned short*)(w + 2826240);  // 512x256 bf16 = 262144 B

  const size_t HC = (size_t)NNODES * RR;

  hipMemcpyAsync(hA, hidden0, HC * sizeof(float), hipMemcpyDeviceToDevice, stream);
  hipMemcpyAsync(cA, cell0,   HC * sizeof(float), hipMemcpyDeviceToDevice, stream);
  k_init<<<(RR * KPAD + 255) / 256, 256, 0, stream>>>(hidden0, hbfT);
  k_wprep<<<(EE * 2048 + 512 * 256 + 255) / 256, 256, 0, stream>>>(W_t, W_ih, W_hh, wtT, wcatT);

  float* h_cur = hA; float* h_nxt = hB;
  float* c_cur = cA; float* c_nxt = cB;

  for (int s = 0; s < SSTEPS; ++s) {
    k_social<<<NNODES, 128, 0, stream>>>(
        grids + (size_t)s * NNODES * NNODES * G2C, hbfT, soc);
    k_dense<<<NNODES / 16, 128, 0, stream>>>(
        soc, input_data + (size_t)s * NNODES * 2,
        h_cur, c_cur, wtT, wcatT,
        W_in, b_in, b_t, b_ih, b_hh, W_out, b_out,
        h_nxt, c_nxt, hbfT,
        out + (size_t)s * NNODES * OUTC);
    float* th = h_cur; h_cur = h_nxt; h_nxt = th;
    float* tc = c_cur; c_cur = c_nxt; c_nxt = tc;
  }

  float* h_fin = out + (size_t)SSTEPS * NNODES * OUTC;
  float* c_fin = h_fin + HC;
  hipMemcpyAsync(h_fin, h_cur, HC * sizeof(float), hipMemcpyDeviceToDevice, stream);
  hipMemcpyAsync(c_fin, c_cur, HC * sizeof(float), hipMemcpyDeviceToDevice, stream);
}
